// MorphologicalLoss_6408091206302
// MI455X (gfx1250) — compile-verified, matrix operand order checked by reading, not run on hardware
//
#include <hip/hip_runtime.h>

// Morphological loss for [32,1,512,512] fp32 pred/target -> scalar fp32.
//
// Roofline: 64 MB compulsory reads ~= 2.8 us at 23.3 TB/s; stencil re-reads are
// L2-resident (64 MB << 192 MB L2). The hot loop is issue-bound, so the math is
// expressed on 2-wide fp32 vectors to lower onto CDNA5's packed VALU
// (v_pk_fma_f32 / v_pk_mul_f32 / v_pk_add_f32, VOP3P): float4 (b128) loads,
// vector-internal left/right neighbors, wave-uniform y-border branches,
// clamp-free morphology (inputs in [0,1) make the clips identities), and
// |dil_p - dil_t| == |Qp - Qt| on complement products. Wave reduction runs on
// the matrix pipe: v_wmma_f32_16x16x32_f16 with B = ones and an exact
// triple-f16 split of each fp32 accumulator (3*11 mantissa bits >= fp32's 24).
// Deterministic two-phase reduction, no atomics.

typedef _Float16 v16h __attribute__((ext_vector_type(16)));
typedef float    v8f  __attribute__((ext_vector_type(8)));
typedef float    v2f  __attribute__((ext_vector_type(2)));

#define BATCH 32
#define HH    512
#define WW    512
#define NPIX  (BATCH * HH * WW)     // 8,388,608
#define NGRP  (NPIX / 4)            // 2,097,152 groups of 4 row-adjacent px
#define GPR   (WW / 4)              // 128 groups per row
#define NBLK  1024
#define NTHR  256

__device__ __forceinline__ v2f vabs2(v2f v) { return __builtin_elementwise_abs(v); }

// Exact wave32 sum via WMMA:
// D[m][*] = x_m + x_{m+16}; sum of d[0..7] on lane 0 covers rows 0-7,
// on lane 16 rows 8-15; total = readlane0 + readlane16.
__device__ __forceinline__ float wave_sum_wmma(float x) {
  _Float16 h1 = (_Float16)x;
  float    r1 = x - (float)h1;
  _Float16 h2 = (_Float16)r1;
  float    r2 = r1 - (float)h2;
  _Float16 h3 = (_Float16)r2;

  v16h a = {};
  a[0] = h1; a[1] = h2; a[2] = h3;
  v16h b;
#pragma unroll
  for (int i = 0; i < 16; ++i) b[i] = (_Float16)1.0f;
  v8f c = {};
  v8f d = __builtin_amdgcn_wmma_f32_16x16x32_f16(false, a, false, b,
                                                 (short)0, c, false, false);
  float r = d[0] + d[1] + d[2] + d[3] + d[4] + d[5] + d[6] + d[7];
  return __int_as_float(__builtin_amdgcn_readlane(__float_as_int(r), 0)) +
         __int_as_float(__builtin_amdgcn_readlane(__float_as_int(r), 16));
}

__device__ __forceinline__ float4 ld4(const float* p) {
  return *reinterpret_cast<const float4*>(p);
}

__global__ __launch_bounds__(NTHR)
void morph_partials(const float* __restrict__ pred,
                    const float* __restrict__ targ,
                    float* __restrict__ partial) {
  v2f S0 = {0.f, 0.f}, S1 = {0.f, 0.f}, S2 = {0.f, 0.f}, S3 = {0.f, 0.f};
  v2f S4 = {0.f, 0.f}, S5 = {0.f, 0.f}, S6 = {0.f, 0.f}, S7 = {0.f, 0.f};

  const unsigned GS = NBLK * NTHR;                 // 262144; NGRP/GS = 8 exactly
  for (unsigned g = blockIdx.x * NTHR + threadIdx.x; g < NGRP; g += GS) {
    const unsigned gx = g & (GPR - 1);             // group column (0..127)
    const unsigned y  = (g >> 7) & (HH - 1);       // row within image (wave-uniform)
    const unsigned i  = g * 4;                     // first pixel of group

    const float4 cp4 = ld4(pred + i);
    const float4 ct4 = ld4(targ + i);

    float4 up4 = make_float4(0.f, 0.f, 0.f, 0.f);
    float4 ut4 = make_float4(0.f, 0.f, 0.f, 0.f);
    float4 dp4 = make_float4(0.f, 0.f, 0.f, 0.f);
    float4 dt4 = make_float4(0.f, 0.f, 0.f, 0.f);
    if (y > 0)      { up4 = ld4(pred + i - WW); ut4 = ld4(targ + i - WW); } // uniform
    if (y < HH - 1) { dp4 = ld4(pred + i + WW); dt4 = ld4(targ + i + WW); } // uniform

    // edge neighbors: clamped address + select-to-zero (branch free)
    const bool hasL = (gx > 0), hasR = (gx < GPR - 1);
    const unsigned jl = i - (hasL ? 1u : 0u);
    const unsigned jr = i + (hasR ? 4u : 3u);
    float pm1 = pred[jl]; pm1 = hasL ? pm1 : 0.f;
    float tm1 = targ[jl]; tm1 = hasL ? tm1 : 0.f;
    float pp4 = pred[jr]; pp4 = hasR ? pp4 : 0.f;
    float tp4 = targ[jr]; tp4 = hasR ? tp4 : 0.f;

    // split the 4-px group into two packed fp32 pairs
    const v2f cpl = {cp4.x, cp4.y}, cph = {cp4.z, cp4.w};
    const v2f ctl = {ct4.x, ct4.y}, cth = {ct4.z, ct4.w};
    const v2f lpl = {pm1,   cp4.x}, lph = {cp4.y, cp4.z};
    const v2f ltl = {tm1,   ct4.x}, lth = {ct4.y, ct4.z};
    const v2f rpl = {cp4.y, cp4.z}, rph = {cp4.w, pp4};
    const v2f rtl = {ct4.y, ct4.z}, rth = {ct4.w, tp4};
    const v2f upl = {up4.x, up4.y}, uph = {up4.z, up4.w};
    const v2f utl = {ut4.x, ut4.y}, uth = {ut4.z, ut4.w};
    const v2f dpl = {dp4.x, dp4.y}, dph = {dp4.z, dp4.w};
    const v2f dtl = {dt4.x, dt4.y}, dth = {dt4.z, dt4.w};

    auto half = [&](v2f p, v2f t, v2f lp, v2f lt, v2f rp, v2f rt,
                    v2f up, v2f ut, v2f dn, v2f dt_, bool firstPair) {
      S0 += vabs2(p - t);                          // L1 recon
      S1 += p * t;                                 // dice intersection (pk_fma)
      S2 += p;
      S3 += t;

      // x-gradient; low-pair .x is invalid at a row start -> masked
      v2f gxv = vabs2(vabs2(p - lp) - vabs2(t - lt));
      if (firstPair) gxv.x = hasL ? gxv.x : 0.f;
      S4 += gxv;

      // erosion: prod over cross (padded zeros already in place)
      const v2f ep = (p * lp) * (rp * up) * dn;
      const v2f et = (t * lt) * (rt * ut) * dt_;
      S7 += vabs2(ep - et);

      // dilation: |dil_p - dil_t| = |Qp - Qt|, Q = prod of complements
      const v2f qp = ((1.f - p) * (1.f - lp)) * ((1.f - rp) * (1.f - up)) * (1.f - dn);
      const v2f qt = ((1.f - t) * (1.f - lt)) * ((1.f - rt) * (1.f - ut)) * (1.f - dt_);
      S6 += vabs2(qp - qt);
    };
    half(cpl, ctl, lpl, ltl, rpl, rtl, upl, utl, dpl, dtl, true);
    half(cph, cth, lph, lth, rph, rth, uph, uth, dph, dth, false);

    if (y > 0) {                                   // wave-uniform y-gradient
      S5 += vabs2(vabs2(cpl - upl) - vabs2(ctl - utl));
      S5 += vabs2(vabs2(cph - uph) - vabs2(cth - uth));
    }
  }

  float s[8] = {S0.x + S0.y, S1.x + S1.y, S2.x + S2.y, S3.x + S3.y,
                S4.x + S4.y, S5.x + S5.y, S6.x + S6.y, S7.x + S7.y};

  __shared__ float wsum[8][8];                     // [wave][acc]
  const unsigned wave = threadIdx.x >> 5;
  const unsigned lane = threadIdx.x & 31;
#pragma unroll
  for (int a = 0; a < 8; ++a) {
    float w = wave_sum_wmma(s[a]);                 // full-EXEC, uniform flow
    if (lane == 0) wsum[wave][a] = w;
  }
  __syncthreads();
  if (threadIdx.x == 0) {
#pragma unroll
    for (int a = 0; a < 8; ++a) {
      float r = 0.f;
      for (int w = 0; w < 8; ++w) r += wsum[w][a];
      partial[blockIdx.x * 8 + a] = r;
    }
  }
}

__global__ __launch_bounds__(NTHR)
void morph_final(const float* __restrict__ partial, float* __restrict__ out) {
  float s[8];
#pragma unroll
  for (int a = 0; a < 8; ++a) {
    float acc = 0.f;
    for (unsigned i = threadIdx.x; i < NBLK; i += NTHR)   // 1024/256 = 4, uniform
      acc += partial[i * 8 + a];
    s[a] = acc;
  }

  __shared__ float wsum[8][8];
  const unsigned wave = threadIdx.x >> 5;
  const unsigned lane = threadIdx.x & 31;
#pragma unroll
  for (int a = 0; a < 8; ++a) {
    float w = wave_sum_wmma(s[a]);
    if (lane == 0) wsum[wave][a] = w;
  }
  __syncthreads();

  if (threadIdx.x == 0) {
    float tot[8];
#pragma unroll
    for (int a = 0; a < 8; ++a) {
      float r = 0.f;
      for (int w = 0; w < 8; ++w) r += wsum[w][a];
      tot[a] = r;
    }
    const float N  = (float)NPIX;                       // 32*512*512
    const float Ng = (float)(BATCH * HH * (WW - 1));    // 32*512*511 (== 32*511*512)

    float total = tot[0] / N;                           // L1 recon
    float dice = (2.f * tot[1] + 1e-5f) / (tot[2] + tot[3] + 1e-5f);
    dice = fminf(fmaxf(dice, 0.f), 1.f);
    total += 0.3f * (-logf(dice + 1e-8f));              // dice term
    total += 0.2f * 0.5f * (tot[4] / Ng + tot[5] / Ng); // boundary term
    total += 0.2f * 0.5f * (tot[6] / N + tot[7] / N);   // morphological term
    out[0] = fmaxf(total, 0.f);
  }
}

extern "C" void kernel_launch(void* const* d_in, const int* in_sizes, int n_in,
                              void* d_out, int out_size, void* d_ws, size_t ws_size,
                              hipStream_t stream) {
  (void)in_sizes; (void)n_in; (void)out_size; (void)ws_size;
  const float* pred = (const float*)d_in[0];
  const float* targ = (const float*)d_in[1];
  float* partial = (float*)d_ws;   // NBLK*8 floats = 32 KB, fully rewritten each call

  morph_partials<<<NBLK, NTHR, 0, stream>>>(pred, targ, partial);
  morph_final<<<1, NTHR, 0, stream>>>(partial, (float*)d_out);
}
